// GraphConv_87342454931924
// MI455X (gfx1250) — compile-verified, matrix operand order checked by reading, not run on hardware
//
#include <hip/hip_runtime.h>

typedef __attribute__((ext_vector_type(16))) _Float16 v16h;
typedef __attribute__((ext_vector_type(8)))  _Float16 v8h;
typedef __attribute__((ext_vector_type(8)))  float    v8f;
typedef __attribute__((ext_vector_type(4)))  float    v4f;

#define NIN  128
#define NOUT 64

// ---------------------------------------------------------------------------
// Zero the output (harness poisons d_out; SpMM accumulates with atomics).
// ---------------------------------------------------------------------------
__global__ __launch_bounds__(256) void gc_zero(float* __restrict__ out, int n4) {
  int i = blockIdx.x * 256 + threadIdx.x;
  if (i < n4) ((float4*)out)[i] = make_float4(0.f, 0.f, 0.f, 0.f);
}

// ---------------------------------------------------------------------------
// Stage 1: h = x @ w via v_wmma_f32_16x16x32_f16.
// One wave computes a 16-row x 64-col strip of h (4 WMMA tiles, K-loop of 4).
// w is staged transposed in LDS as f16 so each B fragment is contiguous.
// ---------------------------------------------------------------------------
__global__ __launch_bounds__(256) void gc_gemm(const float* __restrict__ x,
                                               const float* __restrict__ w,
                                               float* __restrict__ h,
                                               int n_strips) {
  __shared__ _Float16 wT[NOUT][NIN];  // wT[n][k] = (f16) w[k][n]  (16 KB)

  int tid = threadIdx.x;
  for (int i = tid; i < NIN * NOUT; i += 256) {
    int k = i >> 6;    // row of w  (i / 64)
    int n = i & 63;    // col of w
    wT[n][k] = (_Float16)w[i];
  }
  __syncthreads();

  int wave  = tid >> 5;
  int lane  = tid & 31;
  int strip = blockIdx.x * 8 + wave;     // 16-row strip index (wave-uniform)
  if (strip >= n_strips) return;         // uniform per wave: EXEC stays all-1s

  int m16 = lane & 15;   // A: row within strip; B/C/D: column within tile
  int hi  = lane >> 4;   // lane-half selects K sub-chunks / C row offset

  const float* xrow = x + ((long)strip * 16 + m16) * NIN;

  v8f acc[4] = {};       // four 16x16 f32 tiles covering N=0..63

  #pragma unroll
  for (int kt = 0; kt < 4; ++kt) {
    // 16-bit A layout: lanes 0-15 hold K=[k0,k0+8) and [k0+16,k0+24) with k0=kt*32;
    // lanes 16-31 hold the +8 shifted chunks.
    int k0 = kt * 32 + hi * 8;
    v4f a0 = *(const v4f*)(xrow + k0);
    v4f a1 = *(const v4f*)(xrow + k0 + 4);
    v4f a2 = *(const v4f*)(xrow + k0 + 16);
    v4f a3 = *(const v4f*)(xrow + k0 + 20);
    v16h A;
    #pragma unroll
    for (int i = 0; i < 4; ++i) {
      A[i]      = (_Float16)a0[i];
      A[4 + i]  = (_Float16)a1[i];
      A[8 + i]  = (_Float16)a2[i];
      A[12 + i] = (_Float16)a3[i];
    }

    #pragma unroll
    for (int t = 0; t < 4; ++t) {
      int n = t * 16 + m16;  // this lane's B column
      v8h b0 = *(const v8h*)(&wT[n][k0]);        // 128-bit LDS loads
      v8h b1 = *(const v8h*)(&wT[n][k0 + 16]);
      v16h B;
      #pragma unroll
      for (int i = 0; i < 8; ++i) { B[i] = b0[i]; B[8 + i] = b1[i]; }

      acc[t] = __builtin_amdgcn_wmma_f32_16x16x32_f16(
          /*neg_a=*/false, A, /*neg_b=*/false, B,
          /*c_mod=*/(short)0, acc[t], /*reuse_a=*/false, /*reuse_b=*/false);
    }
  }

  // C/D layout: lanes 0-15 -> VGPR j = row M=j; lanes 16-31 -> M=j+8; N = lane&15.
  float* hout = h + (long)strip * 16 * NOUT;
  #pragma unroll
  for (int t = 0; t < 4; ++t) {
    #pragma unroll
    for (int j = 0; j < 8; ++j) {
      hout[(j + hi * 8) * NOUT + t * 16 + m16] = acc[t][j];
    }
  }
}

// ---------------------------------------------------------------------------
// Stage 2: SpMM scatter. One wave32 per edge; each lane owns 2 features.
// h and out are both L2-resident (25.6 MB each vs 192 MB L2).
// ---------------------------------------------------------------------------
__global__ __launch_bounds__(256) void gc_spmm(const float* __restrict__ h,
                                               const int* __restrict__ erow,
                                               const int* __restrict__ ecol,
                                               const float* __restrict__ eval,
                                               float* __restrict__ out,
                                               int n_edges) {
  int e    = (blockIdx.x * 256 + threadIdx.x) >> 5;  // wave-uniform edge id
  int lane = threadIdx.x & 31;
  if (e >= n_edges) return;

  int   r = erow[e];   // same address across the wave -> single L2 request
  int   c = ecol[e];
  float v = eval[e];

  const float2 hv = *(const float2*)(h + (long)c * NOUT + lane * 2);  // 256B/wave, coalesced
  float* o = out + (long)r * NOUT + lane * 2;
  atomicAdd(o,     hv.x * v);   // global_atomic_add_f32 (no return -> STOREcnt)
  atomicAdd(o + 1, hv.y * v);
}

// ---------------------------------------------------------------------------
extern "C" void kernel_launch(void* const* d_in, const int* in_sizes, int n_in,
                              void* d_out, int out_size, void* d_ws, size_t ws_size,
                              hipStream_t stream) {
  const float* x    = (const float*)d_in[0];
  const float* w    = (const float*)d_in[1];
  const int*   erow = (const int*)d_in[2];
  const int*   ecol = (const int*)d_in[3];
  const float* eval = (const float*)d_in[4];
  float* out = (float*)d_out;
  float* h   = (float*)d_ws;               // [n_nodes x 64] f32 scratch (25.6 MB)

  int n_nodes  = in_sizes[0] / NIN;
  int n_edges  = in_sizes[2];
  int n_strips = (n_nodes + 15) / 16;

  int n4 = out_size / 4;                   // out_size = n_nodes*64, divisible by 4
  gc_zero<<<(n4 + 255) / 256, 256, 0, stream>>>(out, n4);

  gc_gemm<<<(n_strips + 7) / 8, 256, 0, stream>>>(x, w, h, n_strips);

  long threads = (long)n_edges * 32;       // one wave per edge
  gc_spmm<<<(int)((threads + 255) / 256), 256, 0, stream>>>(h, erow, ecol, eval,
                                                            out, n_edges);
}
